// A3Cmicropolis10x10linAC_15745350108057
// MI455X (gfx1250) — matrix operand order repaired and checked by reading, not compiled, not run on hardware
//
#include <hip/hip_runtime.h>
#include <hip/hip_bf16.h>

typedef __attribute__((ext_vector_type(16))) __bf16 v16bf;
typedef __attribute__((ext_vector_type(8)))  float  v8f;
typedef __attribute__((ext_vector_type(4)))  int    v4i;

union ABf { v16bf v; uint4 u[2]; };

#define LDS_PTR __attribute__((address_space(3)))
#define GLB_PTR __attribute__((address_space(1)))

static __device__ __forceinline__ float sigmoidf_(float x) {
  return 1.f / (1.f + __expf(-x));
}

static __device__ __forceinline__ void wait_async0() {
#if __has_builtin(__builtin_amdgcn_s_wait_asynccnt)
  __builtin_amdgcn_s_wait_asynccnt(0);
#else
  asm volatile("s_wait_asynccnt 0x0" ::: "memory");
#endif
}

// Cooperative global->LDS stage of `n16` uint4 chunks, async path if available.
static __device__ __forceinline__ void stage_lds(uint4* lds, const uint4* glb,
                                                 int n16) {
#if __has_builtin(__builtin_amdgcn_global_load_async_to_lds_b128)
  for (int i = threadIdx.x; i < n16; i += 256) {
    __builtin_amdgcn_global_load_async_to_lds_b128(
        (GLB_PTR v4i*)(glb + i), (LDS_PTR v4i*)(lds + i), 0, 0);
  }
  wait_async0();
#else
  for (int i = threadIdx.x; i < n16; i += 256) lds[i] = glb[i];
#endif
  __syncthreads();
}

// ---------------------------------------------------------------------------
// Pack FC weight W[N][K] (row-major f32) into WMMA B-operand layout:
// blocks (nb over N/16, kb over K/32), 32 lanes x 16 bf16 per block.
// Element e (0..15) of lane: K = kb*32 + 16*(e>>3) + 8*(lane>>4) + (e&7), N = nb*16 + (lane&15).
// ---------------------------------------------------------------------------
__global__ void pack_fc_w(const float* __restrict__ W, __bf16* __restrict__ P,
                          int N, int K) {
  int KB = K >> 5;
  int total = (N >> 4) * KB * 512;
  for (int tid = blockIdx.x * blockDim.x + threadIdx.x; tid < total;
       tid += gridDim.x * blockDim.x) {
    int e    = tid & 15;
    int lane = (tid >> 4) & 31;
    int blk  = tid >> 9;
    int kb   = blk % KB;
    int nb   = blk / KB;
    int k    = kb * 32 + ((e >> 3) << 4) + ((lane >> 4) << 3) + (e & 7);
    int n    = nb * 16 + (lane & 15);
    P[tid] = (__bf16)W[(size_t)n * K + k];
  }
}

// gates_w [5][64][32][3][3] -> P [5][9 taps][4 ocTiles][32 lanes][16]
__global__ void pack_conv_w(const float* __restrict__ GW, __bf16* __restrict__ P) {
  int total = 5 * 9 * 4 * 512;
  for (int tid = blockIdx.x * blockDim.x + threadIdx.x; tid < total;
       tid += gridDim.x * blockDim.x) {
    int e    = tid & 15;
    int lane = (tid >> 4) & 31;
    int blk  = tid >> 9;              // ((cell*9 + t)*4 + nb)
    int nb   = blk & 3;
    int t    = (blk >> 2) % 9;
    int cell = blk / 36;
    int ic   = ((e >> 3) << 4) + ((lane >> 4) << 3) + (e & 7);        // K
    int oc   = nb * 16 + (lane & 15);         // N
    P[tid] = (__bf16)GW[(((size_t)(cell * 64 + oc)) * 32 + ic) * 9 + t];
  }
}

// conv_in (1x1, 14->16) + ReLU, write bf16 channel-last into xcat[b][p][0..15]
__global__ void conv_in_relu(const float* __restrict__ x,
                             const float* __restrict__ w,
                             const float* __restrict__ bias,
                             __bf16* __restrict__ xcat) {
  int tid = blockIdx.x * blockDim.x + threadIdx.x;
  if (tid >= 4096 * 100) return;
  int b = tid / 100, p = tid % 100;
  const float* xb = x + (size_t)b * 1400 + p;   // x[b][ic][p]
  __bf16* dst = xcat + ((size_t)b * 100 + p) * 32;
  #pragma unroll
  for (int oc = 0; oc < 16; ++oc) {
    float s = bias[oc];
    #pragma unroll
    for (int ic = 0; ic < 14; ++ic) s += w[oc * 14 + ic] * xb[ic * 100];
    dst[oc] = (__bf16)fmaxf(s, 0.f);
  }
}

// hx[idx][b][j][p] f32 -> xcat[b][p][16+j] bf16
__global__ void hx_half(const float* __restrict__ hx_cell,
                        __bf16* __restrict__ xcat) {
  int tid = blockIdx.x * blockDim.x + threadIdx.x;
  if (tid >= 4096 * 1600) return;
  int b = tid / 1600;
  int r = tid % 1600;
  int j = r / 100, p = r % 100;
  xcat[((size_t)b * 100 + p) * 32 + 16 + j] = (__bf16)hx_cell[tid];
}

// ---------------------------------------------------------------------------
// Fused ConvLSTM cell: 3x3 conv (32->64) as 9 tap WMMAs (K=32 each), the
// 4 oc-tiles are exactly the i/r/o/g gates; LSTM update fused in registers.
// One wave = (16-batch tile, one spatial position). gridDim.x = 4096/16.
// ---------------------------------------------------------------------------
__global__ void __launch_bounds__(256, 1)
convlstm_cell(const __bf16* __restrict__ xin, __bf16* __restrict__ xout,
              const __bf16* __restrict__ gwp, const float* __restrict__ gb,
              const float* __restrict__ cxin, float* __restrict__ hsout,
              float* __restrict__ csout) {
  __shared__ uint4 Blds[2304];  // 9 taps * 4 ocTiles * 512 bf16 = 36 KB
  stage_lds(Blds, (const uint4*)gwp, 2304);

  int lane = threadIdx.x & 31;
  int wv = __builtin_amdgcn_readfirstlane((int)(threadIdx.x >> 5));  // scalar
  int lm = lane & 15, lg = lane >> 4;
  int b0 = blockIdx.x * 16;

  float bi0 = gb[lm], bi1 = gb[16 + lm], bi2 = gb[32 + lm], bi3 = gb[48 + lm];
  const __bf16* arow_base = xin + ((size_t)(b0 + lm) * 100) * 32 + lg * 8;
  const uint4* bl = &Blds[lane * 2];

  for (int p = wv; p < 100; p += 8) {  // p is scalar -> uniform tap branches
    v8f zero = {0.f, 0.f, 0.f, 0.f, 0.f, 0.f, 0.f, 0.f};
    v8f acc[4] = {zero, zero, zero, zero};
    int r0 = p / 10, c0 = p % 10;
    #pragma unroll
    for (int t = 0; t < 9; ++t) {
      int r2 = r0 + t / 3 - 1, c2 = c0 + t % 3 - 1;
      if ((unsigned)r2 > 9u || (unsigned)c2 > 9u) continue;  // scalar branch
      int sp = r2 * 10 + c2;
      ABf A;
      const __bf16* ap = arow_base + sp * 32;
      A.u[0] = *(const uint4*)(ap);
      A.u[1] = *(const uint4*)(ap + 16);
      // Preload all 4 B tiles into distinct regs: one ds clause, one wait,
      // then 4 back-to-back WMMAs.
      ABf Bm[4];
      #pragma unroll
      for (int nb = 0; nb < 4; ++nb) {
        Bm[nb].u[0] = bl[(t * 4 + nb) * 64];
        Bm[nb].u[1] = bl[(t * 4 + nb) * 64 + 1];
      }
      #pragma unroll
      for (int nb = 0; nb < 4; ++nb) {
        acc[nb] = __builtin_amdgcn_wmma_f32_16x16x32_bf16(
            false, A.v, false, Bm[nb].v, (short)0, acc[nb], false, false);
      }
    }
    // Epilogue: C layout -> VGPR q holds batch row q + 8*lg, column lm.
    // acc[0]=i, acc[1]=r, acc[2]=o, acc[3]=g (the jnp.split order).
    #pragma unroll
    for (int q = 0; q < 8; ++q) {
      int b = b0 + q + 8 * lg;
      float i_ = sigmoidf_(acc[0][q] + bi0);
      float r_ = sigmoidf_(acc[1][q] + bi1);
      float o_ = sigmoidf_(acc[2][q] + bi2);
      float g_ = tanhf(acc[3][q] + bi3);
      size_t off = ((size_t)b * 16 + lm) * 100 + p;
      float cn = r_ * cxin[off] + i_ * g_;
      float hn = o_ * tanhf(cn);
      csout[off] = cn;
      hsout[off] = hn;
      xout[((size_t)b * 100 + p) * 32 + lm] = (__bf16)hn;
    }
  }
}

// conv_out (3x3, 16->8) + tanh -> y bf16 [B][800] (800 = oc*100 + p)
__global__ void conv_out_tanh(const __bf16* __restrict__ xcat,
                              const float* __restrict__ w,
                              const float* __restrict__ bias,
                              __bf16* __restrict__ y) {
  int tid = blockIdx.x * blockDim.x + threadIdx.x;
  if (tid >= 4096 * 800) return;
  int b = tid / 800;
  int oc = (tid % 800) / 100;
  int p = tid % 100;
  int r0 = p / 10, c0 = p % 10;
  float s = bias[oc];
  for (int kh = 0; kh < 3; ++kh) {
    for (int kw = 0; kw < 3; ++kw) {
      int r2 = r0 + kh - 1, c2 = c0 + kw - 1;
      if ((unsigned)r2 > 9u || (unsigned)c2 > 9u) continue;
      const __bf16* xp = xcat + ((size_t)b * 100 + r2 * 10 + c2) * 32;
      #pragma unroll
      for (int ic = 0; ic < 16; ++ic)
        s += (float)xp[ic] * w[((oc * 16 + ic) * 3 + kh) * 3 + kw];
    }
  }
  y[(size_t)b * 800 + oc * 100 + p] = (__bf16)tanhf(s);
}

// ---------------------------------------------------------------------------
// bf16 WMMA GEMM: C[M,N] = act(A[M,K] * Wp + bias). One wave per 16x16 tile;
// grid = (N/16, M/16/8), 8 waves/block. Whole ntile weight column staged in
// LDS (async) and shared by the 8 waves. KB/N compile-time -> full unroll.
// ---------------------------------------------------------------------------
template <int KB, int NN, bool TANH, bool WF32>
__global__ void __launch_bounds__(256, 1)
gemm_bf16(const __bf16* __restrict__ A, const __bf16* __restrict__ Wp,
          const float* __restrict__ bias, float* __restrict__ Cf,
          __bf16* __restrict__ Cb) {
  __shared__ uint4 Wlds[KB * 64];
  constexpr int K = KB * 32;
  int ntile = blockIdx.x;
  stage_lds(Wlds, (const uint4*)(Wp + (size_t)ntile * KB * 512), KB * 64);

  int lane = threadIdx.x & 31;
  int wv = __builtin_amdgcn_readfirstlane((int)(threadIdx.x >> 5));
  int lm = lane & 15, lg = lane >> 4;
  int mtile = blockIdx.y * 8 + wv;
  const __bf16* arow = A + (size_t)(mtile * 16 + lm) * K + lg * 8;
  const uint4* wl = &Wlds[lane * 2];
  v8f acc = {0.f, 0.f, 0.f, 0.f, 0.f, 0.f, 0.f, 0.f};
  #pragma unroll
  for (int kb = 0; kb < KB; ++kb) {
    ABf Am, Bm;
    Am.u[0] = *(const uint4*)(arow + kb * 32);
    Am.u[1] = *(const uint4*)(arow + kb * 32 + 16);
    Bm.u[0] = wl[kb * 64];
    Bm.u[1] = wl[kb * 64 + 1];
    acc = __builtin_amdgcn_wmma_f32_16x16x32_bf16(
        false, Am.v, false, Bm.v, (short)0, acc, false, false);
  }
  int col = ntile * 16 + lm;
  float bv = bias[col];
  #pragma unroll
  for (int q = 0; q < 8; ++q) {
    int row = mtile * 16 + q + 8 * lg;
    float v = acc[q] + bv;
    if (TANH) v = tanhf(v);
    if (WF32) Cf[(size_t)row * NN + col] = v;
    else      Cb[(size_t)row * NN + col] = (__bf16)v;
  }
}

// value head: [B,800] x [800] + bc -> [B]
__global__ void value_k(const __bf16* __restrict__ y1,
                        const float* __restrict__ wc,
                        const float* __restrict__ bc, float* __restrict__ out) {
  int b = blockIdx.x * blockDim.x + threadIdx.x;
  if (b >= 4096) return;
  float s = bc[0];
  const __bf16* yr = y1 + (size_t)b * 800;
  for (int k = 0; k < 800; ++k) s += (float)yr[k] * wc[k];
  out[b] = s;
}

extern "C" void kernel_launch(void* const* d_in, const int* in_sizes, int n_in,
                              void* d_out, int out_size, void* d_ws,
                              size_t ws_size, hipStream_t stream) {
  (void)in_sizes; (void)n_in; (void)out_size; (void)ws_size;
  const float* x   = (const float*)d_in[0];
  const float* hx  = (const float*)d_in[1];
  const float* cx  = (const float*)d_in[2];
  const float* ciw = (const float*)d_in[3];
  const float* cib = (const float*)d_in[4];
  const float* gw  = (const float*)d_in[5];
  const float* gb  = (const float*)d_in[6];
  const float* cow = (const float*)d_in[7];
  const float* cob = (const float*)d_in[8];
  const float* w0  = (const float*)d_in[9];
  const float* b0  = (const float*)d_in[10];
  const float* w1  = (const float*)d_in[11];
  const float* b1  = (const float*)d_in[12];
  const float* wc  = (const float*)d_in[13];
  const float* bc  = (const float*)d_in[14];
  const float* wa  = (const float*)d_in[15];
  const float* ba  = (const float*)d_in[16];

  char* ws = (char*)d_ws;
  size_t off = 0;
  auto carve = [&](size_t bytes) -> char* {
    char* p = ws + off;
    off += (bytes + 255) & ~(size_t)255;
    return p;
  };
  __bf16* xcatA = (__bf16*)carve((size_t)4096 * 3200 * 2);
  __bf16* xcatB = (__bf16*)carve((size_t)4096 * 3200 * 2);
  __bf16* gwp   = (__bf16*)carve((size_t)5 * 9 * 4 * 512 * 2);
  __bf16* w0p   = (__bf16*)carve((size_t)32 * 25 * 512 * 2);   // N=512,K=800
  __bf16* w1p   = (__bf16*)carve((size_t)50 * 16 * 512 * 2);   // N=800,K=512
  __bf16* wap   = (__bf16*)carve((size_t)50 * 25 * 512 * 2);   // N=800,K=800
  __bf16* ybf   = (__bf16*)carve((size_t)4096 * 800 * 2);
  __bf16* y0    = (__bf16*)carve((size_t)4096 * 512 * 2);
  __bf16* y1    = (__bf16*)carve((size_t)4096 * 800 * 2);

  float* out = (float*)d_out;
  float* val = out;
  float* act = out + 4096;
  float* hs  = act + (size_t)4096 * 800;
  float* cs  = hs + (size_t)5 * 4096 * 1600;

  // Weight packing (tiny)
  pack_fc_w<<<512, 256, 0, stream>>>(w0, w0p, 512, 800);
  pack_fc_w<<<512, 256, 0, stream>>>(w1, w1p, 800, 512);
  pack_fc_w<<<1024, 256, 0, stream>>>(wa, wap, 800, 800);
  pack_conv_w<<<360, 256, 0, stream>>>(gw, gwp);

  // Stage cell-0 input: conv_in -> ch0..15, hx[2] -> ch16..31
  conv_in_relu<<<1600, 256, 0, stream>>>(x, ciw, cib, xcatA);
  static const int ORD[5] = {2, 1, 0, 3, 4};
  hx_half<<<25600, 256, 0, stream>>>(hx + (size_t)ORD[0] * 4096 * 1600, xcatA);

  __bf16* cur = xcatA;
  __bf16* nxt = xcatB;
  for (int k = 0; k < 5; ++k) {
    if (k < 4)
      hx_half<<<25600, 256, 0, stream>>>(hx + (size_t)ORD[k + 1] * 4096 * 1600, nxt);
    convlstm_cell<<<256, 256, 0, stream>>>(
        cur, nxt, gwp + (size_t)k * 9 * 4 * 512, gb + k * 64,
        cx + (size_t)ORD[k] * 4096 * 1600, hs + (size_t)k * 4096 * 1600,
        cs + (size_t)k * 4096 * 1600);
    __bf16* t = cur; cur = nxt; nxt = t;
  }

  conv_out_tanh<<<12800, 256, 0, stream>>>(cur, cow, cob, ybf);

  gemm_bf16<25, 512, true,  false><<<dim3(32, 32), 256, 0, stream>>>(ybf, w0p, b0, nullptr, y0);
  gemm_bf16<16, 800, true,  false><<<dim3(50, 32), 256, 0, stream>>>(y0, w1p, b1, nullptr, y1);
  gemm_bf16<25, 800, false, true ><<<dim3(50, 32), 256, 0, stream>>>(y1, wap, ba, act, nullptr);
  value_k<<<16, 256, 0, stream>>>(y1, wc, bc, val);
}
